// MultiQueryAttention_90417651516239
// MI455X (gfx1250) — matrix operand order repaired and checked by reading, not compiled, not run on hardware
//
#include <hip/hip_runtime.h>
#include <hip/hip_bf16.h>

// ---------------------------------------------------------------------------
// MQA layer for MI455X (gfx1250): bf16 WMMA everywhere, f32 accumulate.
// ---------------------------------------------------------------------------

typedef __attribute__((ext_vector_type(16))) __bf16 v16bf;
typedef __attribute__((ext_vector_type(8)))  __bf16 v8bf;
typedef __attribute__((ext_vector_type(4)))  __bf16 v4bf;
typedef __attribute__((ext_vector_type(8)))  float  v8f;

static constexpr int   B_  = 2;
static constexpr int   S_  = 2048;
static constexpr int   DM_ = 2048;
static constexpr int   H_  = 16;
static constexpr int   KV_ = 4;
static constexpr int   D_  = 128;
static constexpr float MASK_NEG = -2.3819763e+38f;

union U16x8 { uint4 u; __bf16 h[8]; };

// --- gfx1250 async global->LDS copies (guarded: fall back to sync path) ----
#if defined(__has_builtin)
#  if __has_builtin(__builtin_amdgcn_global_load_async_to_lds_b128) && \
      __has_builtin(__builtin_amdgcn_s_wait_asynccnt)
#    define USE_ASYNC_LDS 1
#  endif
#endif
#ifndef USE_ASYNC_LDS
#  define USE_ASYNC_LDS 0
#endif

// builtin signature: void(int4 AS1* src, int4 AS3* dst, imm offset, imm cpol)
typedef int gcc_v4i __attribute__((vector_size(16)));
typedef __attribute__((address_space(1))) gcc_v4i as1_v4i;
typedef __attribute__((address_space(3))) gcc_v4i as3_v4i;

static __device__ inline void copy16_g2l(const __bf16* g, __bf16* l) {
#if USE_ASYNC_LDS
    __builtin_amdgcn_global_load_async_to_lds_b128(
        (as1_v4i*)(uintptr_t)g,
        (as3_v4i*)(unsigned int)(uintptr_t)l,
        0, 0);
#else
    *(uint4*)l = *(const uint4*)g;
#endif
}

static __device__ inline void async_copy_wait() {
#if USE_ASYNC_LDS
    __builtin_amdgcn_s_wait_asynccnt(0);
#endif
}

static __device__ inline v16bf cat16(v8bf lo, v8bf hi) {
    return __builtin_shufflevector(lo, hi, 0,1,2,3,4,5,6,7,8,9,10,11,12,13,14,15);
}

static __device__ inline v8f wmma_bf16(v16bf a, v16bf b, v8f c) {
    // D = A(16x32 bf16) x B(32x16 bf16) + C(16x16 f32)
    return __builtin_amdgcn_wmma_f32_16x16x32_bf16(
        /*neg_a=*/false, a, /*neg_b=*/false, b,
        /*c_mod=*/(short)0, c, /*reuse_a=*/false, /*reuse_b=*/false);
}

static __device__ inline v8f zero8() {
    v8f z = {0.f,0.f,0.f,0.f,0.f,0.f,0.f,0.f};
    return z;
}

// ---------------------------------------------------------------------------
// 1) f32 -> bf16 conversion (vectorized)
// ---------------------------------------------------------------------------
__global__ void f32_to_bf16_kernel(const float* __restrict__ src,
                                   __bf16* __restrict__ dst, int n) {
    int i = (blockIdx.x * blockDim.x + threadIdx.x) * 4;
    if (i + 3 < n) {
        float4 f = *(const float4*)(src + i);
        v4bf o = { (__bf16)f.x, (__bf16)f.y, (__bf16)f.z, (__bf16)f.w };
        *(v4bf*)(dst + i) = o;
    }
}

// ---------------------------------------------------------------------------
// 2) Tiled bf16 WMMA GEMM: C[M,N] = A[M,K] * B[K,N]   (row-major)
//    Block tile 128x128, BK=32, 256 threads = 8 waves, wave tile 64x32.
// ---------------------------------------------------------------------------
template <typename OutT>
__global__ __launch_bounds__(256)
void gemm_bf16_kernel(const __bf16* __restrict__ A, const __bf16* __restrict__ Bm,
                      OutT* __restrict__ C, int Mn, int Nn, int Kn) {
    __shared__ alignas(16) __bf16 As[128][40];   // [row][k]  (BK=32 + pad 8)
    __shared__ alignas(16) __bf16 Bt[128][40];   // [col][k]  transposed B tile

    const int tid  = threadIdx.x;
    const int lane = tid & 31;
    const int wave = tid >> 5;
    const int wr   = wave >> 2;        // 0..1  (row of 64)
    const int wc   = wave & 3;         // 0..3  (col of 32)
    const int mrow = lane & 15;
    const int abase = (lane & 16) >> 1;   // A/P-frag K offset: 0 or 8
    const int bbase = (lane & 16);        // unused (B frag uses abase: 0 or 8)

    const size_t rowBase = (size_t)blockIdx.y * 128;
    const size_t colBase = (size_t)blockIdx.x * 128;

    v8f acc[4][2];
    for (int mt = 0; mt < 4; ++mt)
        for (int nt = 0; nt < 2; ++nt) acc[mt][nt] = zero8();

    for (int k0 = 0; k0 < Kn; k0 += 32) {
        __syncthreads();
        // --- stage A tile 128x32 (2 x b128 per thread, async to LDS) ---
        #pragma unroll
        for (int i = 0; i < 2; ++i) {
            int c    = tid + 256 * i;          // 512 chunks of 8 bf16
            int row  = c >> 2;
            int col8 = (c & 3) * 8;
            copy16_g2l(&A[(rowBase + row) * (size_t)Kn + k0 + col8],
                       &As[row][col8]);
        }
        // --- stage B tile 32x128 transposed into Bt[col][k] ---
        #pragma unroll
        for (int i = 0; i < 2; ++i) {
            int c  = tid + 256 * i;
            int kr = c >> 4;
            int n8 = (c & 15) * 8;
            U16x8 u;
            u.u = *(const uint4*)&Bm[(size_t)(k0 + kr) * Nn + colBase + n8];
            #pragma unroll
            for (int j = 0; j < 8; ++j) Bt[n8 + j][kr] = u.h[j];
        }
        async_copy_wait();
        __syncthreads();

        // --- 8 WMMAs per wave per k-step; B frags hoisted (reused 4x) ---
        v16bf bF[2];
        #pragma unroll
        for (int nt = 0; nt < 2; ++nt) {
            const __bf16* brow = &Bt[wc * 32 + nt * 16 + mrow][abase];
            bF[nt] = cat16(*(const v8bf*)&brow[0], *(const v8bf*)&brow[16]);
        }
        #pragma unroll
        for (int mt = 0; mt < 4; ++mt) {
            const __bf16* arow = &As[wr * 64 + mt * 16 + mrow][0];
            v16bf aF = cat16(*(const v8bf*)&arow[abase],
                             *(const v8bf*)&arow[abase + 16]);
            #pragma unroll
            for (int nt = 0; nt < 2; ++nt)
                acc[mt][nt] = wmma_bf16(aF, bF[nt], acc[mt][nt]);
        }
    }

    // --- write C: element (m = r + 8*(lane>=16), n = lane&15) of each tile ---
    const int hOff = (lane & 16) >> 1;   // 0 or 8
    #pragma unroll
    for (int mt = 0; mt < 4; ++mt)
        #pragma unroll
        for (int nt = 0; nt < 2; ++nt)
            #pragma unroll
            for (int r = 0; r < 8; ++r) {
                size_t row = rowBase + wr * 64 + mt * 16 + r + hOff;
                size_t col = colBase + wc * 32 + nt * 16 + mrow;
                C[row * (size_t)Nn + col] = (OutT)acc[mt][nt][r];
            }
}

// ---------------------------------------------------------------------------
// 3) RoPE + scatter: xqkv[B*S,3072] -> Q[B,H,S,D], K[B,KV,S,D], V[B,KV,S,D]
// ---------------------------------------------------------------------------
__global__ __launch_bounds__(256)
void rope_scatter_kernel(const __bf16* __restrict__ xqkv,
                         __bf16* __restrict__ Q, __bf16* __restrict__ K,
                         __bf16* __restrict__ V) {
    const int bs = blockIdx.x;           // 0 .. B*S-1
    const int b  = bs / S_;
    const int s  = bs % S_;
    const __bf16* row = xqkv + (size_t)bs * (H_ + 2 * KV_) * D_;
    const int tid = threadIdx.x;

    // rotary halves for q (16 heads) and k (4 heads): (H+KV)*64 pairs
    for (int idx = tid; idx < (H_ + KV_) * 64; idx += 256) {
        int head = idx >> 6;
        int d    = idx & 63;
        float freq = __powf(10000.f, -(float)d / 64.f);
        float sn, cs;
        __sincosf((float)s * freq, &sn, &cs);
        if (head < H_) {
            float x1 = (float)row[head * D_ + d];
            float x2 = (float)row[head * D_ + d + 64];
            __bf16* dst = Q + (((size_t)b * H_ + head) * S_ + s) * D_;
            dst[d]      = (__bf16)(x1 * cs - x2 * sn);
            dst[d + 64] = (__bf16)(x1 * sn + x2 * cs);
        } else {
            int kh = head - H_;
            float x1 = (float)row[H_ * D_ + kh * D_ + d];
            float x2 = (float)row[H_ * D_ + kh * D_ + d + 64];
            __bf16* dst = K + (((size_t)b * KV_ + kh) * S_ + s) * D_;
            dst[d]      = (__bf16)(x1 * cs - x2 * sn);
            dst[d + 64] = (__bf16)(x1 * sn + x2 * cs);
        }
    }
    // v pass-through
    for (int idx = tid; idx < KV_ * D_; idx += 256) {
        int kh = idx >> 7;
        int d  = idx & 127;
        V[(((size_t)b * KV_ + kh) * S_ + s) * D_ + d] =
            row[(H_ + KV_) * D_ + kh * D_ + d];
    }
}

// ---------------------------------------------------------------------------
// 4) Causal flash attention (bf16 WMMA, online softmax)
//    grid = (S/64, B*H), block = 128 (4 waves). Wave w owns 16 query rows.
// ---------------------------------------------------------------------------
__global__ __launch_bounds__(128)
void flash_attn_kernel(const __bf16* __restrict__ Qg,
                       const __bf16* __restrict__ Kg,
                       const __bf16* __restrict__ Vg,
                       __bf16* __restrict__ ATT) {
    __shared__ alignas(16) __bf16 Qs[64][136];    // 64 q-rows x 128 d
    __shared__ alignas(16) __bf16 Ks[32][136];    // 32 keys  x 128 d (row-major)
    __shared__ alignas(16) __bf16 Vt[128][40];    // 128 d    x 32 keys (transposed)
    __shared__ alignas(16) __bf16 Ps[4][16][40];  // per-wave P tile 16x32

    const int tid  = threadIdx.x;
    const int lane = tid & 31;
    const int w    = tid >> 5;
    const int mrow = lane & 15;
    const int abase = (lane & 16) >> 1;  // 0/8
    const int bbase = (lane & 16);       // 0/16
    const int hOff  = (lane & 16) >> 1;  // 0/8 row offset in C tiles

    const int bh    = blockIdx.y;
    const int b     = bh / H_;
    const int h     = bh % H_;
    const int kvh   = h / (H_ / KV_);
    const int qbase = blockIdx.x * 64;

    const __bf16* Qp = Qg + (((size_t)b * H_  + h  ) * S_ + qbase) * D_;
    const __bf16* Kp = Kg + (((size_t)b * KV_ + kvh) * S_) * D_;
    const __bf16* Vp = Vg + (((size_t)b * KV_ + kvh) * S_) * D_;

    // stage Q tile: 64x128 -> 1024 chunks of 8 bf16, 8 per thread (async)
    #pragma unroll
    for (int i = 0; i < 8; ++i) {
        int c    = tid + 128 * i;
        int row  = c >> 4;
        int col8 = (c & 15) * 8;
        copy16_g2l(&Qp[(size_t)row * D_ + col8], &Qs[row][col8]);
    }
    async_copy_wait();

    v8f o[8];
    #pragma unroll
    for (int dt = 0; dt < 8; ++dt) o[dt] = zero8();
    float mrun[8], lrun[8];
    #pragma unroll
    for (int r = 0; r < 8; ++r) { mrun[r] = -3.0e38f; lrun[r] = 0.f; }

    const float scale = 0.08838834764831845f;   // 1/sqrt(128)
    const int   nkv   = (qbase + 64) / 32;      // causal tile count

    for (int kt = 0; kt < nkv; ++kt) {
        __syncthreads();
        // stage K tile (async, row-major) and V tile (transposed), 4 chunks each
        #pragma unroll
        for (int i = 0; i < 4; ++i) {
            int c    = tid + 128 * i;
            int row  = c >> 4;
            int col8 = (c & 15) * 8;
            copy16_g2l(&Kp[((size_t)kt * 32 + row) * D_ + col8], &Ks[row][col8]);
            U16x8 u;
            u.u = *(const uint4*)&Vp[((size_t)kt * 32 + row) * D_ + col8];
            #pragma unroll
            for (int j = 0; j < 8; ++j) Vt[col8 + j][row] = u.h[j];
        }
        if (kt + 1 < nkv)
            __builtin_prefetch(&Kp[((size_t)(kt + 1) * 32) * D_], 0, 0);
        async_copy_wait();
        __syncthreads();

        // --- scores S(16x32) = Q(16x128) . K^T : 8 WMMAs ---
        v8f s0 = zero8(), s1 = zero8();
        #pragma unroll
        for (int kc = 0; kc < 4; ++kc) {
            const __bf16* qrow = &Qs[w * 16 + mrow][kc * 32];
            v16bf aF = cat16(*(const v8bf*)&qrow[abase],
                             *(const v8bf*)&qrow[abase + 16]);
            const __bf16* k0 = &Ks[mrow][kc * 32 + abase];
            v16bf b0 = cat16(*(const v8bf*)&k0[0], *(const v8bf*)&k0[16]);
            s0 = wmma_bf16(aF, b0, s0);
            const __bf16* k1 = &Ks[16 + mrow][kc * 32 + abase];
            v16bf b1 = cat16(*(const v8bf*)&k1[0], *(const v8bf*)&k1[16]);
            s1 = wmma_bf16(aF, b1, s1);
        }

        // --- scale + causal mask + online softmax (per row r) ---
        const int kg0 = kt * 32 + mrow;
        const int kg1 = kg0 + 16;
        #pragma unroll
        for (int r = 0; r < 8; ++r) {
            const int mg = qbase + w * 16 + r + hOff;
            float v0 = s0[r] * scale; if (kg0 > mg) v0 = MASK_NEG;
            float v1 = s1[r] * scale; if (kg1 > mg) v1 = MASK_NEG;
            float mx = fmaxf(v0, v1);
            #pragma unroll
            for (int d = 1; d < 16; d <<= 1)
                mx = fmaxf(mx, __shfl_xor(mx, d, 32));
            float mnew  = fmaxf(mrun[r], mx);
            float alpha = __expf(mrun[r] - mnew);
            mrun[r] = mnew;
            float p0 = __expf(v0 - mnew);
            float p1 = __expf(v1 - mnew);
            float rs = p0 + p1;
            #pragma unroll
            for (int d = 1; d < 16; d <<= 1)
                rs += __shfl_xor(rs, d, 32);
            lrun[r] = lrun[r] * alpha + rs;
            #pragma unroll
            for (int dt = 0; dt < 8; ++dt) o[dt][r] *= alpha;
            // stash P as bf16 in per-wave LDS tile [row][key]
            Ps[w][r + hOff][mrow]      = (__bf16)p0;
            Ps[w][r + hOff][mrow + 16] = (__bf16)p1;
        }

        // --- O(16x128) += P(16x32) . V(32x128) : 8 WMMAs ---
        const __bf16* prow = &Ps[w][mrow][0];
        v16bf pF = cat16(*(const v8bf*)&prow[abase],
                         *(const v8bf*)&prow[abase + 16]);
        #pragma unroll
        for (int dt = 0; dt < 8; ++dt) {
            const __bf16* vrow = &Vt[dt * 16 + mrow][abase];
            v16bf bF = cat16(*(const v8bf*)&vrow[0], *(const v8bf*)&vrow[16]);
            o[dt] = wmma_bf16(pF, bF, o[dt]);
        }
    }

    // --- normalize and write ATT[b, s, h*D + d] ---
    float inv[8];
    #pragma unroll
    for (int r = 0; r < 8; ++r) inv[r] = 1.f / lrun[r];
    #pragma unroll
    for (int dt = 0; dt < 8; ++dt)
        #pragma unroll
        for (int r = 0; r < 8; ++r) {
            int srow = qbase + w * 16 + r + hOff;
            int d    = dt * 16 + mrow;
            ATT[((size_t)b * S_ + srow) * (size_t)(H_ * D_) + h * D_ + d] =
                (__bf16)(o[dt][r] * inv[r]);
        }
}

// ---------------------------------------------------------------------------
// launcher
// ---------------------------------------------------------------------------
extern "C" void kernel_launch(void* const* d_in, const int* in_sizes, int n_in,
                              void* d_out, int out_size, void* d_ws, size_t ws_size,
                              hipStream_t stream) {
    (void)in_sizes; (void)n_in; (void)out_size; (void)ws_size;

    const float* x    = (const float*)d_in[0];   // [B,S,DM]
    const float* Wqkv = (const float*)d_in[1];   // [DM, (H+2KV)*D]
    const float* Wo   = (const float*)d_in[2];   // [H*D, DM]
    float* out        = (float*)d_out;           // [B,S,DM]

    const size_t nX    = (size_t)B_ * S_ * DM_;             // 8388608
    const size_t nWqkv = (size_t)DM_ * (H_ + 2 * KV_) * D_; // 6291456
    const size_t nWo   = (size_t)(H_ * D_) * DM_;           // 4194304
    const size_t nQKV  = (size_t)B_ * S_ * (H_ + 2 * KV_) * D_;
    const size_t nQ    = (size_t)B_ * H_  * S_ * D_;
    const size_t nK    = (size_t)B_ * KV_ * S_ * D_;
    const size_t nATT  = (size_t)B_ * S_ * H_ * D_;

    char* ws = (char*)d_ws;
    size_t off = 0;
    __bf16* Xb    = (__bf16*)(ws + off); off += nX    * 2;
    __bf16* Wqkvb = (__bf16*)(ws + off); off += nWqkv * 2;
    __bf16* Wob   = (__bf16*)(ws + off); off += nWo   * 2;
    __bf16* XQKV  = (__bf16*)(ws + off); off += nQKV  * 2;
    __bf16* Qb    = (__bf16*)(ws + off); off += nQ    * 2;
    __bf16* Kb    = (__bf16*)(ws + off); off += nK    * 2;
    __bf16* Vb    = (__bf16*)(ws + off); off += nK    * 2;
    __bf16* ATTb  = (__bf16*)(ws + off); off += nATT  * 2;

    // 1) convert inputs to bf16
    f32_to_bf16_kernel<<<dim3((unsigned)(nX    / 1024)), 256, 0, stream>>>(x,    Xb,    (int)nX);
    f32_to_bf16_kernel<<<dim3((unsigned)(nWqkv / 1024)), 256, 0, stream>>>(Wqkv, Wqkvb, (int)nWqkv);
    f32_to_bf16_kernel<<<dim3((unsigned)(nWo   / 1024)), 256, 0, stream>>>(Wo,   Wob,   (int)nWo);

    // 2) xqkv = x @ Wqkv   (4096 x 3072 x 2048)
    gemm_bf16_kernel<__bf16><<<dim3((H_ + 2 * KV_) * D_ / 128, B_ * S_ / 128),
                               256, 0, stream>>>(Xb, Wqkvb, XQKV,
                                                 B_ * S_, (H_ + 2 * KV_) * D_, DM_);

    // 3) RoPE + scatter into attention layouts
    rope_scatter_kernel<<<dim3(B_ * S_), 256, 0, stream>>>(XQKV, Qb, Kb, Vb);

    // 4) causal flash attention
    flash_attn_kernel<<<dim3(S_ / 64, B_ * H_), 128, 0, stream>>>(Qb, Kb, Vb, ATTb);

    // 5) out = att @ Wo   (4096 x 2048 x 2048), f32 output
    gemm_bf16_kernel<float><<<dim3(DM_ / 128, B_ * S_ / 128),
                              256, 0, stream>>>(ATTb, Wob, out,
                                                B_ * S_, DM_, H_ * D_);
}
